// NGCFModel_45835890983575
// MI455X (gfx1250) — matrix operand order repaired and checked by reading, not compiled, not run on hardware
//
#include <hip/hip_runtime.h>

// ---------------------------------------------------------------------------
// NGCF propagate_embeddings for MI455X (gfx1250, wave32).
//   x (node embeddings per layer) : bf16  [N,64]   (halves SpMM gather bytes)
//   side (A@x accumulator)        : f32   [N,64]   (f32 atomics, L2-resident)
//   dense part: v_wmma_f32_16x16x32_bf16, one wave per 16-row tile
// ---------------------------------------------------------------------------

#define KDIM 64

typedef __attribute__((ext_vector_type(16))) __bf16          v16bf;
typedef __attribute__((ext_vector_type(16))) unsigned short  v16u;
typedef __attribute__((ext_vector_type(8)))  float           v8f;

__device__ __forceinline__ unsigned short f2bf(float f) {
  union { float f; unsigned u; } v; v.f = f;
  unsigned u = v.u;
  u += 0x7FFFu + ((u >> 16) & 1u);           // round-to-nearest-even
  return (unsigned short)(u >> 16);
}
__device__ __forceinline__ float bf2f(unsigned short b) {
  union { unsigned u; float f; } v; v.u = ((unsigned)b) << 16;
  return v.f;
}

// ---- layer-0: e = l2norm(concat(Gu,Gi)) -> bf16 x, g_sum = rowsum(e) -------
__global__ void ngcf_prep_kernel(const float* __restrict__ Gu,
                                 const float* __restrict__ Gi,
                                 unsigned short* __restrict__ x,
                                 float* __restrict__ gsum, int U, int N) {
  int wid  = (blockIdx.x * blockDim.x + threadIdx.x) >> 5;  // one wave per row
  int lane = threadIdx.x & 31;
  if (wid >= N) return;
  const float* src = (wid < U) ? (Gu + (size_t)wid * KDIM)
                               : (Gi + (size_t)(wid - U) * KDIM);
  float2 v = *(const float2*)(src + lane * 2);
  float ss = v.x * v.x + v.y * v.y;
#pragma unroll
  for (int m = 16; m >= 1; m >>= 1) ss += __shfl_xor(ss, m, 32);
  float inv = 1.0f / fmaxf(sqrtf(ss), 1e-12f);
  float a = v.x * inv, b = v.y * inv;
  float rs = a + b;
#pragma unroll
  for (int m = 16; m >= 1; m >>= 1) rs += __shfl_xor(rs, m, 32);
  unsigned pack = (unsigned)f2bf(a) | ((unsigned)f2bf(b) << 16);
  *(unsigned*)(x + (size_t)wid * KDIM + lane * 2) = pack;
  if (lane == 0) gsum[wid] = rs;
}

// ---- pre-pack W1/W2 (f32 row-major) into bf16 WMMA B-fragment order --------
// frag(kb,nt): lane l (h=l>>4, n=l&15), elem j = W[kb*32 + h*8 + j + (j>=8 ? 8 : 0)][nt*16 + n]
__global__ void ngcf_wconv_kernel(const float* __restrict__ W1,
                                  const float* __restrict__ W2,
                                  unsigned short* __restrict__ wf, int total) {
  int t = blockIdx.x * blockDim.x + threadIdx.x;
  if (t >= total) return;
  int lane = t & 31, kbnt = (t >> 5) & 7, mtx = (t >> 8) & 1, l = t >> 9;
  int kb = kbnt >> 2, nt = kbnt & 3;
  int h = lane >> 4, n = lane & 15;
  const float* W = (mtx ? W2 : W1) + (size_t)l * KDIM * KDIM;
  unsigned short* dst = wf + (((size_t)(l * 2 + mtx) * 8 + kbnt) * 512 + lane * 16);
#pragma unroll
  for (int j = 0; j < 16; ++j) {
    int k = kb * 32 + h * 8 + j + (j >= 8 ? 8 : 0);
    dst[j] = f2bf(W[k * KDIM + nt * 16 + n]);
  }
}

// ---- SpMM scatter: side[r] += x[c] for both directions of each edge --------
__global__ void ngcf_spmm_kernel(const int* __restrict__ eu,
                                 const int* __restrict__ ei,
                                 const unsigned short* __restrict__ x,
                                 float* __restrict__ side, int E) {
  int w    = (blockIdx.x * blockDim.x + threadIdx.x) >> 5;  // wave per edge
  int lane = threadIdx.x & 31;
  if (w >= E) return;
  int u = eu[w], i = ei[w];
  unsigned pu = *(const unsigned*)(x + (size_t)u * KDIM + lane * 2);
  unsigned pi = *(const unsigned*)(x + (size_t)i * KDIM + lane * 2);
  float* su = side + (size_t)u * KDIM + lane * 2;
  float* si = side + (size_t)i * KDIM + lane * 2;
  atomicAdd(si + 0, bf2f((unsigned short)(pu & 0xFFFFu)));
  atomicAdd(si + 1, bf2f((unsigned short)(pu >> 16)));
  atomicAdd(su + 0, bf2f((unsigned short)(pi & 0xFFFFu)));
  atomicAdd(su + 1, bf2f((unsigned short)(pi >> 16)));
}

// ---- dense layer: h=(side+x)@W1 + (side*x)@W2 + b, leaky, l2norm, in-place -
__global__ void __launch_bounds__(256) ngcf_layer_kernel(
    unsigned short* __restrict__ x, const float* __restrict__ side,
    const unsigned short* __restrict__ wf1, const unsigned short* __restrict__ wf2,
    const float* __restrict__ b1, const float* __restrict__ b2,
    float* __restrict__ gsum, int ntiles) {
  int wid  = (blockIdx.x * blockDim.x + threadIdx.x) >> 5;  // wave per 16 rows
  int lane = threadIdx.x & 31;
  if (wid >= ntiles) return;                 // uniform per wave -> EXEC all-1s
  int r0 = wid * 16;
  int m = lane & 15, h = lane >> 4;

  // Build A fragments: a = side + x, p = side * x (bf16), kb = 0,1 (K=32 each)
  v16u aa[2], mm[2];
  const unsigned short* xr = x + (size_t)(r0 + m) * KDIM;
  const float*          sr = side + (size_t)(r0 + m) * KDIM;
#pragma unroll
  for (int kb = 0; kb < 2; ++kb) {
    int k0 = kb * 32 + h * 8;
    v16u av, mv;
#pragma unroll
    for (int g = 0; g < 2; ++g) {            // elems j<8 -> k0.. ; j>=8 -> k0+16..
      int k = k0 + g * 16;
      uint4  xq = *(const uint4*)(xr + k);   // 8 bf16
      float4 sa = *(const float4*)(sr + k);
      float4 sb = *(const float4*)(sr + k + 4);
      unsigned xw[4] = {xq.x, xq.y, xq.z, xq.w};
      float    sf[8] = {sa.x, sa.y, sa.z, sa.w, sb.x, sb.y, sb.z, sb.w};
#pragma unroll
      for (int j = 0; j < 4; ++j) {
        float x0 = bf2f((unsigned short)(xw[j] & 0xFFFFu));
        float x1 = bf2f((unsigned short)(xw[j] >> 16));
        float s0 = sf[2 * j], s1 = sf[2 * j + 1];
        av[g * 8 + 2 * j]     = f2bf(s0 + x0);
        av[g * 8 + 2 * j + 1] = f2bf(s1 + x1);
        mv[g * 8 + 2 * j]     = f2bf(s0 * x0);
        mv[g * 8 + 2 * j + 1] = f2bf(s1 * x1);
      }
    }
    aa[kb] = av; mm[kb] = mv;
  }

  // 4 n-tiles x (2 + 2) WMMA, accumulate in f32, init with bias
  v8f acc[4];
#pragma unroll
  for (int nt = 0; nt < 4; ++nt) {
    float bias = b1[nt * 16 + m] + b2[nt * 16 + m];
    v8f c;
#pragma unroll
    for (int r = 0; r < 8; ++r) c[r] = bias;
#pragma unroll
    for (int kb = 0; kb < 2; ++kb) {
      v16bf bf = *(const v16bf*)(wf1 + ((size_t)(kb * 4 + nt) * 512 + lane * 16));
      c = __builtin_amdgcn_wmma_f32_16x16x32_bf16(
          false, __builtin_bit_cast(v16bf, aa[kb]), false, bf, (short)0, c, false, false);
    }
#pragma unroll
    for (int kb = 0; kb < 2; ++kb) {
      v16bf bf = *(const v16bf*)(wf2 + ((size_t)(kb * 4 + nt) * 512 + lane * 16));
      c = __builtin_amdgcn_wmma_f32_16x16x32_bf16(
          false, __builtin_bit_cast(v16bf, mm[kb]), false, bf, (short)0, c, false, false);
    }
    acc[nt] = c;
  }

  // leaky-ReLU + row l2-norm (row M = r0 + 8h + r; 16 lanes per half share a row)
  float invn[8];
#pragma unroll
  for (int r = 0; r < 8; ++r) {
    float ss = 0.f;
#pragma unroll
    for (int nt = 0; nt < 4; ++nt) {
      float v = acc[nt][r];
      v = (v >= 0.f) ? v : 0.01f * v;
      acc[nt][r] = v;
      ss += v * v;
    }
    ss += __shfl_xor(ss, 1, 32);
    ss += __shfl_xor(ss, 2, 32);
    ss += __shfl_xor(ss, 4, 32);
    ss += __shfl_xor(ss, 8, 32);
    invn[r] = 1.0f / fmaxf(sqrtf(ss), 1e-12f);
  }
#pragma unroll
  for (int r = 0; r < 8; ++r) {
    int row = r0 + h * 8 + r;
    float rs = 0.f;
#pragma unroll
    for (int nt = 0; nt < 4; ++nt) {
      float v = acc[nt][r] * invn[r];
      rs += v;
      x[(size_t)row * KDIM + nt * 16 + m] = f2bf(v);   // in-place next-layer x
    }
    rs += __shfl_xor(rs, 1, 32);
    rs += __shfl_xor(rs, 2, 32);
    rs += __shfl_xor(rs, 4, 32);
    rs += __shfl_xor(rs, 8, 32);
    if (m == 0) gsum[row] += rs;      // this wave exclusively owns these rows
  }
}

__global__ void ngcf_final_kernel(const float* __restrict__ gsum,
                                  float* __restrict__ out, int N, float scale) {
  int i = blockIdx.x * blockDim.x + threadIdx.x;
  if (i < N) out[i] = gsum[i] * scale;
}

// ---------------------------------------------------------------------------
extern "C" void kernel_launch(void* const* d_in, const int* in_sizes, int n_in,
                              void* d_out, int out_size, void* d_ws, size_t ws_size,
                              hipStream_t stream) {
  const float* Gu = (const float*)d_in[0];
  const float* Gi = (const float*)d_in[1];
  const float* W1 = (const float*)d_in[2];
  const float* b1 = (const float*)d_in[3];
  const float* W2 = (const float*)d_in[4];
  const float* b2 = (const float*)d_in[5];
  const int* eidx = (const int*)d_in[6];

  const int U = in_sizes[0] / KDIM;
  const int I = in_sizes[1] / KDIM;
  const int N = U + I;
  const int L = in_sizes[3] / KDIM;      // b1 is [L,K]
  const int E = in_sizes[6] / 2;
  const int* eu = eidx;
  const int* ei = eidx + E;

  // workspace carve-out (x: bf16 N*64, side: f32 N*64, gsum: f32 N, wfrag)
  char* ws = (char*)d_ws;
  size_t off = 0;
  unsigned short* xbuf = (unsigned short*)(ws + off);
  off += (size_t)N * KDIM * sizeof(unsigned short);
  off = (off + 255) & ~(size_t)255;
  float* side = (float*)(ws + off);
  off += (size_t)N * KDIM * sizeof(float);
  off = (off + 255) & ~(size_t)255;
  float* gsum = (float*)(ws + off);
  off += (size_t)N * sizeof(float);
  off = (off + 255) & ~(size_t)255;
  unsigned short* wfrag = (unsigned short*)(ws + off);   // L*2*4096 bf16

  const int wtot = L * 2 * 8 * 32;
  ngcf_wconv_kernel<<<(wtot + 255) / 256, 256, 0, stream>>>(W1, W2, wfrag, wtot);
  ngcf_prep_kernel<<<(N + 7) / 8, 256, 0, stream>>>(Gu, Gi, xbuf, gsum, U, N);

  const int ntiles = N / 16;             // N = 120000 -> 7500 tiles
  for (int l = 0; l < L; ++l) {
    hipMemsetAsync(side, 0, (size_t)N * KDIM * sizeof(float), stream);
    ngcf_spmm_kernel<<<(E + 7) / 8, 256, 0, stream>>>(eu, ei, xbuf, side, E);
    ngcf_layer_kernel<<<(ntiles + 7) / 8, 256, 0, stream>>>(
        xbuf, side,
        wfrag + (size_t)(l * 2 + 0) * 4096,
        wfrag + (size_t)(l * 2 + 1) * 4096,
        b1 + l * KDIM, b2 + l * KDIM, gsum, ntiles);
  }

  const float scale = 1.0f / (float)((L + 1) * KDIM);   // mean over (L+1)*K
  ngcf_final_kernel<<<(N + 255) / 256, 256, 0, stream>>>(gsum, (float*)d_out, N, scale);
}
